// MultiHeadSelfAttention_26113401160337
// MI455X (gfx1250) — matrix operand order repaired and checked by reading, not compiled, not run on hardware
//
#include <hip/hip_runtime.h>

// ---------------------------------------------------------------------------
// MI455X (gfx1250) multi-head causal self-attention, bf16 WMMA everywhere.
// B=2, T=4096, C=768, H=12, D=64.
// Round 2: 32x64 macro-tiles in the projection GEMMs (1.5 loads/wmma),
//          64-key blocks in flash attention (halves softmax shuffle overhead
//          per wmma).
// ---------------------------------------------------------------------------

#define BB   2
#define TT   4096
#define CC   768
#define HH   12
#define DD   64

typedef __attribute__((ext_vector_type(16))) __bf16 v16bf;
typedef __attribute__((ext_vector_type(8)))  __bf16 v8bf;
typedef __attribute__((ext_vector_type(8)))  float  v8f;

static __device__ __forceinline__ unsigned short f32_to_bf16_bits(float f) {
    unsigned int u = __builtin_bit_cast(unsigned int, f);
    u += 0x7FFFu + ((u >> 16) & 1u);          // round-to-nearest-even
    return (unsigned short)(u >> 16);
}

// A-matrix fragment (16x32 bf16): lanes 0-15 hold K=0..15 of row (lane&15),
// lanes 16-31 hold K=16..31.  Two contiguous 16B chunks per lane.
static __device__ __forceinline__ v16bf load_a_frag(const unsigned short* rowbase,
                                                    int k0, int lane) {
    const int kb = k0 + ((lane & 16) ? 16 : 0);
    union { v16bf v; v8bf h[2]; } u;
    u.h[0] = *reinterpret_cast<const v8bf*>(rowbase + kb);
    u.h[1] = *reinterpret_cast<const v8bf*>(rowbase + kb + 8);
    return u.v;
}

// B-matrix fragment (32x16 bf16): lane holds 16 contiguous K values of column
// n = lane&15 (lanes 0-15: K=0..15, lanes 16-31: K=16..31). One 32B chunk.
static __device__ __forceinline__ v16bf load_b_frag(const unsigned short* rowbase,
                                                    int k0, int lane) {
    const int kb = k0 + ((lane & 16) ? 16 : 0);
    return *reinterpret_cast<const v16bf*>(rowbase + kb);
}

static __device__ __forceinline__ v8f wmma_bf16(v16bf a, v16bf b, v8f c) {
    return __builtin_amdgcn_wmma_f32_16x16x32_bf16(false, a, false, b,
                                                   (short)0, c, false, false);
}

// ---------------------------------------------------------------------------
// 1) fp32 -> bf16 cast
// ---------------------------------------------------------------------------
__global__ void cast_bf16_kernel(const float* __restrict__ in,
                                 unsigned short* __restrict__ out, int n) {
    int i = blockIdx.x * blockDim.x + threadIdx.x;
    if (i < n) out[i] = f32_to_bf16_bits(in[i]);
}

// ---------------------------------------------------------------------------
// 2) Q/K/V projection:  Y = X(bf16, BT x C) @ W(bf16, C x C)^T
//    Each wave computes a 32x64 macro-tile (2 M-subtiles x 4 N-subtiles).
//    mode 0: write (B,H,T,D)   (for Q and K)
//    mode 1: write (B,H,D,T)   (V transposed, feeds P@V B-fragments)
// ---------------------------------------------------------------------------
__global__ void __launch_bounds__(128)
proj_qkv_kernel(const unsigned short* __restrict__ X,
                const unsigned short* __restrict__ W,
                unsigned short* __restrict__ Y, int mode) {
    const int lane  = threadIdx.x & 31;
    const int wave  = threadIdx.x >> 5;
    const int tm0   = blockIdx.x * 2;                  // first 16-row tile of BT
    const int tn0   = (blockIdx.y * 4 + wave) * 4;     // first 16-col tile of C
    const int n16   = lane & 15;
    const int mbase = (lane & 16) ? 8 : 0;

    const unsigned short* xrow0 = X + (size_t)(tm0 * 16 + n16) * CC;
    const unsigned short* xrow1 = xrow0 + (size_t)16 * CC;
    const unsigned short* wrow0 = W + (size_t)(tn0 * 16 + n16) * CC;

    v8f acc[2][4];
#pragma unroll
    for (int i = 0; i < 2; ++i)
#pragma unroll
        for (int j = 0; j < 4; ++j)
            acc[i][j] = v8f{0.f,0.f,0.f,0.f,0.f,0.f,0.f,0.f};

#pragma unroll 2
    for (int k0 = 0; k0 < CC; k0 += 32) {
        const v16bf a0 = load_a_frag(xrow0, k0, lane);
        const v16bf a1 = load_a_frag(xrow1, k0, lane);
#pragma unroll
        for (int j = 0; j < 4; ++j) {
            const v16bf b = load_b_frag(wrow0 + (size_t)j * 16 * CC, k0, lane);
            acc[0][j] = wmma_bf16(a0, b, acc[0][j]);
            acc[1][j] = wmma_bf16(a1, b, acc[1][j]);
        }
    }

    const int hh = tn0 >> 2;           // 64-col supertile aligned: head index
#pragma unroll
    for (int i = 0; i < 2; ++i) {
#pragma unroll
        for (int j = 0; j < 4; ++j) {
            const int d = j * 16 + n16;
            if (mode == 0) {                          // (B,H,T,D)
#pragma unroll
                for (int v = 0; v < 8; ++v) {
                    const int tg = (tm0 + i) * 16 + mbase + v;
                    const int bb = tg >> 12, t = tg & (TT - 1);
                    Y[((size_t)((bb * HH + hh) * TT + t) << 6) + d] =
                        f32_to_bf16_bits(acc[i][j][v]);
                }
            } else {                                  // (B,H,D,T): 8 consecutive t
                const int tg0 = (tm0 + i) * 16 + mbase;
                const int bb  = tg0 >> 12, t0 = tg0 & (TT - 1);
                unsigned short* dst =
                    Y + ((size_t)((bb * HH + hh) * DD + d) << 12) + t0;
                union { v8bf v; unsigned short s[8]; } pk;
#pragma unroll
                for (int v = 0; v < 8; ++v) pk.s[v] = f32_to_bf16_bits(acc[i][j][v]);
                *reinterpret_cast<v8bf*>(dst) = pk.v;
            }
        }
    }
}

// ---------------------------------------------------------------------------
// 3) Causal flash attention.  One wave per 16-query tile; 64 keys/iteration.
//    Q,K: (B,H,T,64) bf16;  Vt: (B,H,64,T) bf16;  CTX out: (B,T,C) bf16.
// ---------------------------------------------------------------------------
#define AT_WPB   4
#define P_STRIDE 72            // 64 keys + 8 pad (rows stay 16B aligned)

__global__ void __launch_bounds__(128)
attn_kernel(const unsigned short* __restrict__ Q,
            const unsigned short* __restrict__ K,
            const unsigned short* __restrict__ Vt,
            unsigned short* __restrict__ CTX) {
    __shared__ __align__(16) unsigned short plds[AT_WPB][16 * P_STRIDE];

    const int lane  = threadIdx.x & 31;
    const int wave  = threadIdx.x >> 5;
    const int qt    = blockIdx.x * AT_WPB + wave;    // 16-query tile
    const int h     = blockIdx.y;
    const int b     = blockIdx.z;
    const int bh    = b * HH + h;
    const int n16   = lane & 15;
    const int mbase = (lane & 16) ? 8 : 0;

    // Q A-fragments (d = 0..31, 32..63), loaded once.
    const unsigned short* qrow = Q + ((size_t)((bh << 12) + qt * 16 + n16) << 6);
    const v16bf aq0 = load_a_frag(qrow, 0, lane);
    const v16bf aq1 = load_a_frag(qrow, 32, lane);

    float m[8], l[8];
#pragma unroll
    for (int v = 0; v < 8; ++v) { m[v] = -1e30f; l[v] = 0.f; }
    v8f o[4];
#pragma unroll
    for (int dt = 0; dt < 4; ++dt) o[dt] = v8f{0.f,0.f,0.f,0.f,0.f,0.f,0.f,0.f};

    const float sc = 0.125f * 1.4426950408889634f;   // 1/sqrt(64) * log2(e)
    const int kmax = qt * 16 + 16;                   // exclusive causal bound
    unsigned short* myp = plds[wave];

    for (int k0 = 0; k0 < kmax; k0 += 64) {
        // --- S = Q @ K^T for 64 keys (four 16-key subtiles) ----------------
        v8f s[4];
#pragma unroll
        for (int st = 0; st < 4; ++st) {
            const unsigned short* krow =
                K + ((size_t)((bh << 12) + k0 + st * 16 + n16) << 6);
            v8f acc = {0.f,0.f,0.f,0.f,0.f,0.f,0.f,0.f};
            acc = wmma_bf16(aq0, load_b_frag(krow, 0,  lane), acc);
            acc = wmma_bf16(aq1, load_b_frag(krow, 32, lane), acc);
            s[st] = acc;
        }

#pragma unroll
        for (int st = 0; st < 4; ++st)
#pragma unroll
            for (int v = 0; v < 8; ++v) s[st][v] *= sc;

        if (k0 + 64 > qt * 16) {                     // blocks touching diagonal
#pragma unroll
            for (int st = 0; st < 4; ++st) {
                const int key = k0 + st * 16 + n16;
#pragma unroll
                for (int v = 0; v < 8; ++v) {
                    const int row = qt * 16 + mbase + v;
                    if (key > row) s[st][v] = -1e30f;
                }
            }
        }

        // --- online softmax (row = half-wave x VGPR index) -----------------
        float alpha[8];
#pragma unroll
        for (int v = 0; v < 8; ++v) {
            float rmx = fmaxf(fmaxf(s[0][v], s[1][v]), fmaxf(s[2][v], s[3][v]));
#pragma unroll
            for (int mk = 1; mk < 16; mk <<= 1)
                rmx = fmaxf(rmx, __shfl_xor(rmx, mk, 32));
            const float nm = fmaxf(m[v], rmx);
            alpha[v] = __builtin_amdgcn_exp2f(m[v] - nm);
            m[v] = nm;
            float rs = 0.f;
#pragma unroll
            for (int st = 0; st < 4; ++st) {
                s[st][v] = __builtin_amdgcn_exp2f(s[st][v] - nm);
                rs += s[st][v];
            }
#pragma unroll
            for (int mk = 1; mk < 16; mk <<= 1)
                rs += __shfl_xor(rs, mk, 32);
            l[v] = l[v] * alpha[v] + rs;
        }

        // --- P (f32, D-layout) -> LDS -> bf16 A-fragments ------------------
#pragma unroll
        for (int st = 0; st < 4; ++st)
#pragma unroll
            for (int v = 0; v < 8; ++v)
                myp[(mbase + v) * P_STRIDE + st * 16 + n16] =
                    f32_to_bf16_bits(s[st][v]);

        const v16bf ap0 = load_a_frag(myp + n16 * P_STRIDE, 0,  lane);
        const v16bf ap1 = load_a_frag(myp + n16 * P_STRIDE, 32, lane);

        // --- O = O*alpha + P @ V ------------------------------------------
#pragma unroll
        for (int dt = 0; dt < 4; ++dt) {
#pragma unroll
            for (int v = 0; v < 8; ++v) o[dt][v] *= alpha[v];
            const unsigned short* vrow =
                Vt + ((size_t)(bh * DD + dt * 16 + n16) << 12);
            o[dt] = wmma_bf16(ap0, load_b_frag(vrow, k0,      lane), o[dt]);
            o[dt] = wmma_bf16(ap1, load_b_frag(vrow, k0 + 32, lane), o[dt]);
        }
    }

    // --- normalize and store ctx (B,T,C) bf16 -----------------------------
#pragma unroll
    for (int v = 0; v < 8; ++v) {
        const float inv = 1.0f / l[v];
        const int t = qt * 16 + mbase + v;
#pragma unroll
        for (int dt = 0; dt < 4; ++dt) {
            CTX[(size_t)((b << 12) + t) * CC + h * DD + dt * 16 + n16] =
                f32_to_bf16_bits(o[dt][v] * inv);
        }
    }
}

// ---------------------------------------------------------------------------
// 4) Output projection: out(f32, BT x C) = CTX(bf16) @ Wo(bf16)^T
//    Same 32x64 macro-tile per wave, fp32 stores.
// ---------------------------------------------------------------------------
__global__ void __launch_bounds__(128)
proj_out_kernel(const unsigned short* __restrict__ X,
                const unsigned short* __restrict__ W,
                float* __restrict__ Y) {
    const int lane  = threadIdx.x & 31;
    const int wave  = threadIdx.x >> 5;
    const int tm0   = blockIdx.x * 2;
    const int tn0   = (blockIdx.y * 4 + wave) * 4;
    const int n16   = lane & 15;
    const int mbase = (lane & 16) ? 8 : 0;

    const unsigned short* xrow0 = X + (size_t)(tm0 * 16 + n16) * CC;
    const unsigned short* xrow1 = xrow0 + (size_t)16 * CC;
    const unsigned short* wrow0 = W + (size_t)(tn0 * 16 + n16) * CC;

    v8f acc[2][4];
#pragma unroll
    for (int i = 0; i < 2; ++i)
#pragma unroll
        for (int j = 0; j < 4; ++j)
            acc[i][j] = v8f{0.f,0.f,0.f,0.f,0.f,0.f,0.f,0.f};

#pragma unroll 2
    for (int k0 = 0; k0 < CC; k0 += 32) {
        const v16bf a0 = load_a_frag(xrow0, k0, lane);
        const v16bf a1 = load_a_frag(xrow1, k0, lane);
#pragma unroll
        for (int j = 0; j < 4; ++j) {
            const v16bf b = load_b_frag(wrow0 + (size_t)j * 16 * CC, k0, lane);
            acc[0][j] = wmma_bf16(a0, b, acc[0][j]);
            acc[1][j] = wmma_bf16(a1, b, acc[1][j]);
        }
    }

#pragma unroll
    for (int i = 0; i < 2; ++i)
#pragma unroll
        for (int j = 0; j < 4; ++j)
#pragma unroll
            for (int v = 0; v < 8; ++v)
                Y[(size_t)((tm0 + i) * 16 + mbase + v) * CC + (tn0 + j) * 16 + n16] =
                    acc[i][j][v];
}

// ---------------------------------------------------------------------------
// Launch
// ---------------------------------------------------------------------------
extern "C" void kernel_launch(void* const* d_in, const int* in_sizes, int n_in,
                              void* d_out, int out_size, void* d_ws, size_t ws_size,
                              hipStream_t stream) {
    const float* x  = (const float*)d_in[0];
    const float* wq = (const float*)d_in[1];
    const float* wk = (const float*)d_in[2];
    const float* wv = (const float*)d_in[3];
    const float* wo = (const float*)d_in[4];
    float* out = (float*)d_out;

    const size_t NX = (size_t)BB * TT * CC;   // 6291456
    const size_t NW = (size_t)CC * CC;        // 589824

    char* ws = (char*)d_ws;
    unsigned short* xb  = (unsigned short*)ws;  ws += NX * 2;
    unsigned short* wqb = (unsigned short*)ws;  ws += NW * 2;
    unsigned short* wkb = (unsigned short*)ws;  ws += NW * 2;
    unsigned short* wvb = (unsigned short*)ws;  ws += NW * 2;
    unsigned short* wob = (unsigned short*)ws;  ws += NW * 2;
    unsigned short* qh  = (unsigned short*)ws;  ws += NX * 2;   // (B,H,T,D)
    unsigned short* kh  = (unsigned short*)ws;  ws += NX * 2;   // (B,H,T,D)
    unsigned short* vt  = (unsigned short*)ws;  ws += NX * 2;   // (B,H,D,T)
    unsigned short* ctx = (unsigned short*)ws;  ws += NX * 2;   // (B,T,C)

    cast_bf16_kernel<<<dim3((NX + 255) / 256), dim3(256), 0, stream>>>(x,  xb,  (int)NX);
    cast_bf16_kernel<<<dim3((NW + 255) / 256), dim3(256), 0, stream>>>(wq, wqb, (int)NW);
    cast_bf16_kernel<<<dim3((NW + 255) / 256), dim3(256), 0, stream>>>(wk, wkb, (int)NW);
    cast_bf16_kernel<<<dim3((NW + 255) / 256), dim3(256), 0, stream>>>(wv, wvb, (int)NW);
    cast_bf16_kernel<<<dim3((NW + 255) / 256), dim3(256), 0, stream>>>(wo, wob, (int)NW);

    // 32 rows x 64 cols per wave, 4 waves/block: grid (8192/32, 768/(64*4))
    dim3 pg(BB * TT / 32, CC / 256);
    proj_qkv_kernel<<<pg, dim3(128), 0, stream>>>(xb, wqb, qh, 0);
    proj_qkv_kernel<<<pg, dim3(128), 0, stream>>>(xb, wkb, kh, 0);
    proj_qkv_kernel<<<pg, dim3(128), 0, stream>>>(xb, wvb, vt, 1);

    dim3 ag(TT / 16 / AT_WPB, HH, BB);    // (64, 12, 2)
    attn_kernel<<<ag, dim3(32 * AT_WPB), 0, stream>>>(qh, kh, vt, ctx);

    proj_out_kernel<<<pg, dim3(128), 0, stream>>>(ctx, wob, out);
}
